// EdgeGNN_40888088658022
// MI455X (gfx1250) — matrix operand order repaired and checked by reading, not compiled, not run on hardware
//
#include <hip/hip_runtime.h>
#include <hip/hip_bf16.h>

// ---------------------------------------------------------------------------
// EdgeConv GNN, restructured:
//   h = relu( x @ (Wtop - Wbot) + neighbor_mean(x) @ Wbot + b )   (0 if deg==0)
//   out = x@Wf0 + h1@Wf1 + h2@Wf2 + h3@Wf3 + bf
// GEMMs on v_wmma_f32_16x16x32_bf16, B staged through LDS in fragment layout.
// ---------------------------------------------------------------------------

typedef __attribute__((ext_vector_type(16))) __bf16       v16bf;
typedef __attribute__((ext_vector_type(8)))  float        v8f;
typedef __attribute__((ext_vector_type(4)))  unsigned int u32x4;

#define DF 256   // feature width for every GEMM operand

struct QPair { u32x4 lo, hi; };

static __device__ inline v16bf frag_from(u32x4 lo, u32x4 hi) {
  QPair p; p.lo = lo; p.hi = hi;
  return __builtin_bit_cast(v16bf, p);
}

// ---- A fragment: 16x32 bf16, row-major source [M,256] --------------------
// ISA layout: lane m=L%16; VGPR0-3 hold K=(L>=16?8:0)+0..7, VGPR4-7 hold +16.
static __device__ inline v16bf load_afrag(const __bf16* __restrict__ X, int M,
                                          int R0, int kk, int lane) {
  int r = R0 + (lane & 15);
  if (r >= M) r = M - 1;                       // clamp; stores are guarded
  int off = (lane >> 4) << 3;                  // 0 or 8 (K elements)
  const __bf16* base = X + (size_t)r * DF + kk + off;
  u32x4 lo = *reinterpret_cast<const u32x4*>(base);        // K = kk+off .. +7
  u32x4 hi = *reinterpret_cast<const u32x4*>(base + 16);   // K = kk+16+off ..
  return frag_from(lo, hi);
}

// ---------------------------------------------------------------------------
// WMMA GEMM: out[M,256] = act( sum_m X_m[M,256] @ W_m^T + bias ), masked.
// Block = 256 threads (8 waves). Block tile 128x256, wave tile 32x128.
// B slab (32k x 256n = 16KB bf16) staged in LDS in WMMA fragment layout:
//   chunk c = ntile*32 + lane  ->  32 bytes = lane's v16bf for that B frag.
// ---------------------------------------------------------------------------
__global__ __launch_bounds__(256) void k_gemm_wmma(
    const __bf16* __restrict__ X0, const __bf16* __restrict__ Wt0,
    const __bf16* __restrict__ X1, const __bf16* __restrict__ Wt1,
    const __bf16* __restrict__ X2, const __bf16* __restrict__ Wt2,
    const __bf16* __restrict__ X3, const __bf16* __restrict__ Wt3,
    int nmat, const float* __restrict__ bias, int M,
    float* __restrict__ outF, __bf16* __restrict__ outB,
    const float* __restrict__ cmask, int relu)
{
  __shared__ __align__(32) unsigned char ldsB[16 * 32 * 32];   // 16 KB

  const int lane  = threadIdx.x & 31;
  const int wave  = threadIdx.x >> 5;
  const int msub  = wave & 3;                  // 4 M-subtiles of 32 rows
  const int nhalf = wave >> 2;                 // 2 N-halves of 128 cols
  const int nbase = nhalf * 128;
  const int R0    = blockIdx.x * 128 + msub * 32;

  v8f acc[2][8];
#pragma unroll
  for (int g = 0; g < 2; ++g)
#pragma unroll
    for (int i = 0; i < 8; ++i)
      acc[g][i] = v8f{0.f, 0.f, 0.f, 0.f, 0.f, 0.f, 0.f, 0.f};

  for (int m = 0; m < nmat; ++m) {
    const __bf16* X  = (m == 0) ? X0 : (m == 1) ? X1 : (m == 2) ? X2 : X3;
    const __bf16* Wt = (m == 0) ? Wt0 : (m == 1) ? Wt1 : (m == 2) ? Wt2 : Wt3;
    __builtin_prefetch(Wt, 0, 1);              // global_prefetch_b8

    for (int kk = 0; kk < DF; kk += 32) {
      // ---- cooperative fill of the 16KB B slab (all 16 ntiles) ----
      __syncthreads();                          // WAR vs previous iteration
#pragma unroll
      for (int c = threadIdx.x; c < 512; c += 256) {
        int nt = c >> 5, l = c & 31;
        int n = nt * 16 + (l & 15);
        int koff = (l >> 4) << 3;               // 0 or 8
        const __bf16* s = Wt + (size_t)n * DF + kk + koff;
        u32x4* d = reinterpret_cast<u32x4*>(ldsB + c * 32);
        d[0] = *reinterpret_cast<const u32x4*>(s);
        d[1] = *reinterpret_cast<const u32x4*>(s + 16);
      }
      __syncthreads();

      // ---- two A fragments (rows R0..+15, R0+16..+31), 8 B frags ----
      v16bf a0 = load_afrag(X, M, R0, kk, lane);
      v16bf a1 = load_afrag(X, M, R0 + 16, kk, lane);
#pragma unroll
      for (int nt = 0; nt < 8; ++nt) {
        const u32x4* bp = reinterpret_cast<const u32x4*>(
            ldsB + (((nhalf * 8 + nt) * 32 + lane) * 32));
        v16bf b = frag_from(bp[0], bp[1]);       // ds_load_b128 x2
        acc[0][nt] = __builtin_amdgcn_wmma_f32_16x16x32_bf16(
            false, a0, false, b, (short)0, acc[0][nt], false, false);
        acc[1][nt] = __builtin_amdgcn_wmma_f32_16x16x32_bf16(
            false, a1, false, b, (short)0, acc[1][nt], false, false);
      }
    }
  }

  // C/D layout: lane n=L%16; VGPR v holds row m = v + 8*(L>=16)
  const int nlo = lane & 15;
  const int mhi = (lane >> 4) * 8;
#pragma unroll
  for (int nt = 0; nt < 8; ++nt) {
    int col = nbase + nt * 16 + nlo;
    float bv = bias ? bias[col] : 0.0f;
#pragma unroll
    for (int g = 0; g < 2; ++g) {
#pragma unroll
      for (int v = 0; v < 8; ++v) {
        int row = R0 + g * 16 + mhi + v;
        if (row < M) {
          float val = acc[g][nt][v] + bv;
          if (relu) val = fmaxf(val, 0.0f);
          if (cmask && cmask[row] < 0.5f) val = 0.0f;  // deg==0 -> exact 0
          outF[(size_t)row * DF + col] = val;
          if (outB) outB[(size_t)row * DF + col] = (__bf16)val;
        }
      }
    }
  }
}

// ---------------------------------------------------------------------------
// Support kernels
// ---------------------------------------------------------------------------
__global__ void k_zero(float* p, int n) {
  int i = blockIdx.x * blockDim.x + threadIdx.x;
  if (i < n) p[i] = 0.0f;
}

__global__ void k_degree(const int* __restrict__ dst, int E, float* __restrict__ cnt) {
  int e = blockIdx.x * blockDim.x + threadIdx.x;
  if (e < E) atomicAdd(&cnt[dst[e]], 1.0f);
}

// one thread per (edge, 4-float chunk): float4 gather + 4 L2 atomic adds
__global__ void k_scatter(const int* __restrict__ src, const int* __restrict__ dst,
                          int E, const float* __restrict__ feat,
                          float* __restrict__ agg) {
  int t = blockIdx.x * blockDim.x + threadIdx.x;
  int e = t >> 6;
  if (e >= E) return;
  int c = (t & 63) << 2;
  float4 v = *reinterpret_cast<const float4*>(feat + (size_t)src[e] * DF + c);
  float* a = agg + (size_t)dst[e] * DF + c;
  atomicAdd(a + 0, v.x);
  atomicAdd(a + 1, v.y);
  atomicAdd(a + 2, v.z);
  atomicAdd(a + 3, v.w);
}

__global__ void k_mean_cvt(const float* __restrict__ agg, const float* __restrict__ cnt,
                           __bf16* __restrict__ agg_b, int total) {
  int i = blockIdx.x * blockDim.x + threadIdx.x;
  if (i >= total) return;
  float c = fmaxf(cnt[i >> 8], 1.0f);
  agg_b[i] = (__bf16)(agg[i] / c);
}

__global__ void k_cvt(const float* __restrict__ in, __bf16* __restrict__ out, int n) {
  int i = blockIdx.x * blockDim.x + threadIdx.x;
  if (i < n) out[i] = (__bf16)in[i];
}

// A = Wtop - Wbot, B = Wbot; stored transposed (n-major) in bf16
__global__ void k_prep_edge_w(const float* __restrict__ W,
                              __bf16* __restrict__ At, __bf16* __restrict__ Bt) {
  int i = blockIdx.x * blockDim.x + threadIdx.x;
  if (i >= DF * DF) return;
  int n = i >> 8, k = i & 255;
  float wt = W[k * DF + n];
  float wb = W[(DF + k) * DF + n];
  At[n * DF + k] = (__bf16)(wt - wb);
  Bt[n * DF + k] = (__bf16)wb;
}

// Wf [1024,256] -> 4 transposed bf16 blocks [256,256]
__global__ void k_prep_final_w(const float* __restrict__ Wf, __bf16* __restrict__ Wft) {
  int i = blockIdx.x * blockDim.x + threadIdx.x;
  if (i >= 4 * DF * DF) return;
  int b = i >> 16, r = i & 65535, n = r >> 8, k = r & 255;
  Wft[(size_t)b * DF * DF + n * DF + k] = (__bf16)Wf[(size_t)(b * DF + k) * DF + n];
}

// ---------------------------------------------------------------------------
extern "C" void kernel_launch(void* const* d_in, const int* in_sizes, int n_in,
                              void* d_out, int out_size, void* d_ws, size_t ws_size,
                              hipStream_t stream) {
  (void)n_in; (void)out_size; (void)ws_size;
  const float* x  = (const float*)d_in[0];
  const int*   ei = (const int*)d_in[1];
  const float* W0 = (const float*)d_in[2];
  const float* b0 = (const float*)d_in[3];
  const float* W1 = (const float*)d_in[4];
  const float* b1 = (const float*)d_in[5];
  const float* W2 = (const float*)d_in[6];
  const float* b2 = (const float*)d_in[7];
  const float* Wf = (const float*)d_in[8];
  const float* bf = (const float*)d_in[9];
  float* out = (float*)d_out;

  const int N = in_sizes[0] / DF;
  const int E = in_sizes[1] / 2;
  const int* src = ei;
  const int* dst = ei + E;

  // workspace carve-up (256B aligned)
  char* p = (char*)d_ws;
  auto carve = [&](size_t bytes) -> char* {
    char* r = p;
    p += (bytes + 255) & ~(size_t)255;
    return r;
  };
  const size_t NF = (size_t)N * DF;
  float*  cnt   = (float*)carve((size_t)N * 4);
  float*  agg   = (float*)carve(NF * 4);
  float*  hA    = (float*)carve(NF * 4);
  float*  hB    = (float*)carve(NF * 4);
  __bf16* x_b   = (__bf16*)carve(NF * 2);
  __bf16* agg_b = (__bf16*)carve(NF * 2);
  __bf16* h1_b  = (__bf16*)carve(NF * 2);
  __bf16* h2_b  = (__bf16*)carve(NF * 2);
  __bf16* h3_b  = (__bf16*)carve(NF * 2);
  __bf16* At0 = (__bf16*)carve(DF * DF * 2);
  __bf16* Bt0 = (__bf16*)carve(DF * DF * 2);
  __bf16* At1 = (__bf16*)carve(DF * DF * 2);
  __bf16* Bt1 = (__bf16*)carve(DF * DF * 2);
  __bf16* At2 = (__bf16*)carve(DF * DF * 2);
  __bf16* Bt2 = (__bf16*)carve(DF * DF * 2);
  __bf16* Wft = (__bf16*)carve((size_t)4 * DF * DF * 2);

  const int T = 256;
  const int gW   = (DF * DF + T - 1) / T;
  const int gWf  = (4 * DF * DF + T - 1) / T;
  const int gNF  = (int)((NF + T - 1) / T);
  const int gN   = (N + T - 1) / T;
  const int gE   = (E + T - 1) / T;
  const int gSc  = (int)(((size_t)E * 64 + T - 1) / T);
  const int gMM  = (N + 127) / 128;

  // weight prep + input conversion + degrees (graph is static across layers)
  k_prep_edge_w<<<gW, T, 0, stream>>>(W0, At0, Bt0);
  k_prep_edge_w<<<gW, T, 0, stream>>>(W1, At1, Bt1);
  k_prep_edge_w<<<gW, T, 0, stream>>>(W2, At2, Bt2);
  k_prep_final_w<<<gWf, T, 0, stream>>>(Wf, Wft);
  k_cvt<<<gNF, T, 0, stream>>>(x, x_b, (int)NF);
  k_zero<<<gN, T, 0, stream>>>(cnt, N);
  k_degree<<<gE, T, 0, stream>>>(dst, E, cnt);

  // layer 1: feat = x
  k_zero<<<gNF, T, 0, stream>>>(agg, (int)NF);
  k_scatter<<<gSc, T, 0, stream>>>(src, dst, E, x, agg);
  k_mean_cvt<<<gNF, T, 0, stream>>>(agg, cnt, agg_b, (int)NF);
  k_gemm_wmma<<<gMM, T, 0, stream>>>(x_b, At0, agg_b, Bt0, nullptr, nullptr,
                                     nullptr, nullptr, 2, b0, N, hA, h1_b, cnt, 1);
  // layer 2: feat = hA
  k_zero<<<gNF, T, 0, stream>>>(agg, (int)NF);
  k_scatter<<<gSc, T, 0, stream>>>(src, dst, E, hA, agg);
  k_mean_cvt<<<gNF, T, 0, stream>>>(agg, cnt, agg_b, (int)NF);
  k_gemm_wmma<<<gMM, T, 0, stream>>>(h1_b, At1, agg_b, Bt1, nullptr, nullptr,
                                     nullptr, nullptr, 2, b1, N, hB, h2_b, cnt, 1);
  // layer 3: feat = hB
  k_zero<<<gNF, T, 0, stream>>>(agg, (int)NF);
  k_scatter<<<gSc, T, 0, stream>>>(src, dst, E, hB, agg);
  k_mean_cvt<<<gNF, T, 0, stream>>>(agg, cnt, agg_b, (int)NF);
  k_gemm_wmma<<<gMM, T, 0, stream>>>(h2_b, At2, agg_b, Bt2, nullptr, nullptr,
                                     nullptr, nullptr, 2, b2, N, hA, h3_b, cnt, 1);

  // final: cat([x,h1,h2,h3]) @ Wf + bf  ==  four K=256 blocks
  k_gemm_wmma<<<gMM, T, 0, stream>>>(x_b, Wft, h1_b, Wft + DF * DF,
                                     h2_b, Wft + 2 * DF * DF, h3_b, Wft + 3 * DF * DF,
                                     4, bf, N, out, nullptr, nullptr, 0);
}
